// MLSTMCharLMLayer_64089501990972
// MI455X (gfx1250) — matrix operand order repaired and checked by reading, not compiled, not run on hardware
//
#include <hip/hip_runtime.h>
#include <hip/hip_bf16.h>

// ---------------------------------------------------------------------------
// mLSTM char-LM layer for MI455X (gfx1250), wave32 + v_wmma_f32_16x16x32_bf16.
// Strategy: weight-norm + fp32->bf16 conversion once (weights then fit in the
// 192MB L2: wh 128MB + wmh 32MB), then 64 recurrence steps stream weights from
// L2 with bf16 WMMA / fp32 accumulation. Recurrence is launched as per-step
// kernels to honor the sequential dependence.
// ---------------------------------------------------------------------------

typedef __bf16 bf16;
typedef __attribute__((ext_vector_type(16))) __bf16 v16bf;
typedef __attribute__((ext_vector_type(8)))  __bf16 v8bf;
typedef __attribute__((ext_vector_type(8)))  float  v8f;

#define B_SZ   32
#define T_SZ   64
#define NEMBD  64
#define NVOCAB 256
#define HID    4096
#define ZDIM   (4 * HID)

// ---- WMMA fragment helpers (wave32, 16x16x32 bf16) -------------------------
// A: 16x32 (MxK). Lane l<16: row M=l, K = 0..15; lane l>=16: row
// M=l-16, K = 16..31 (matches B fragment K order).
static __device__ __forceinline__ v16bf load_a_frag(const bf16* A, int lda,
                                                    int m0, int k0) {
  const int l  = threadIdx.x & 31;
  const int m  = m0 + (l & 15);
  const int kb = k0 + ((l >> 4) << 4); // +0 or +16
  const bf16* p = A + (size_t)m * lda + kb;
  v8bf lo = *(const v8bf*)p;        // K = kb .. kb+7
  v8bf hi = *(const v8bf*)(p + 8);  // K = kb+8 .. kb+15
  v16bf r;
#pragma unroll
  for (int i = 0; i < 8; ++i) { r[i] = lo[i]; r[i + 8] = hi[i]; }
  return r;
}

// B: 32x16 (KxN), stored transposed as BT[n][k] (N-major) so each lane pulls
// one contiguous 32B chunk. Lane l<16: col N=l, K=0..15; lane l>=16: K=16..31.
static __device__ __forceinline__ v16bf load_b_frag(const bf16* BT, int ldb,
                                                    int n0, int k0) {
  const int l  = threadIdx.x & 31;
  const int n  = n0 + (l & 15);
  const int kk = k0 + ((l >> 4) << 4); // +0 or +16
  return *(const v16bf*)(BT + (size_t)n * ldb + kk);
}

static __device__ __forceinline__ v8f wmma_bf16(v16bf a, v16bf b, v8f c) {
  return __builtin_amdgcn_wmma_f32_16x16x32_bf16(false, a, false, b,
                                                 (short)0, c, false, false);
}

// ---- prep: column-wise weight norm + transpose + bf16 convert --------------
// w is [K, N] row-major; out is [N, K] bf16. scale_n = g[n]/sqrt(max(ss,eps)).
__global__ void wn_transpose_kernel(const float* __restrict__ w,
                                    const float* __restrict__ g,
                                    bf16* __restrict__ outT,
                                    int K, int N, int has_g) {
  const int n = blockIdx.x * blockDim.x + threadIdx.x;
  if (n >= N) return;
  float scale = 1.0f;
  if (has_g) {
    float ss = 0.f;
    for (int k = 0; k < K; ++k) {
      float v = w[(size_t)k * N + n];
      ss += v * v;
    }
    scale = g[n] * rsqrtf(fmaxf(ss, 1e-12f));
  }
  for (int k = 0; k < K; ++k)
    outT[(size_t)n * K + k] = (bf16)(w[(size_t)k * N + n] * scale);
}

// ---- prep: embedding gather to bf16, [T, B, NEMBD] -------------------------
__global__ void embed_kernel(const int* __restrict__ X,
                             const float* __restrict__ embd,
                             bf16* __restrict__ words) {
  const int idx = blockIdx.x * blockDim.x + threadIdx.x;
  if (idx >= T_SZ * B_SZ * NEMBD) return;
  const int e = idx % NEMBD;
  const int b = (idx / NEMBD) % B_SZ;
  const int t = idx / (NEMBD * B_SZ);
  const int tok = X[b * T_SZ + t];
  words[idx] = (bf16)embd[(size_t)tok * NEMBD + e];
}

// ---- prep: initial state from S[2,B,HID] -----------------------------------
__global__ void init_state_kernel(const float* __restrict__ S,
                                  float* __restrict__ c_ws,
                                  float* __restrict__ h_ws,
                                  bf16* __restrict__ h_bf) {
  const int idx = blockIdx.x * blockDim.x + threadIdx.x;
  if (idx >= B_SZ * HID) return;
  c_ws[idx] = S[idx];
  const float h = S[(size_t)B_SZ * HID + idx];
  h_ws[idx] = h;
  h_bf[idx] = (bf16)h;
}

// ---- per-step kernel 1: m = (x @ wmx) * (h @ wmh), output bf16 [32,HID] ----
// Block: 256 thr = 8 waves; tile 32 rows x 64 cols; wave -> (m_tile, n_sub).
__global__ __launch_bounds__(256)
void step_m_kernel(const bf16* __restrict__ words, const bf16* __restrict__ h_bf,
                   const bf16* __restrict__ wmxT, const bf16* __restrict__ wmhT,
                   bf16* __restrict__ m_bf, int t) {
  const int wave = threadIdx.x >> 5;
  const int l    = threadIdx.x & 31;
  const int m0   = (wave & 1) * 16;
  const int n0   = blockIdx.x * 64 + (wave >> 1) * 16;
  const bf16* x  = words + (size_t)t * B_SZ * NEMBD;

  // h @ wmh, 2-way K-split for independent WMMA chains
  v8f acc2a = {}, acc2b = {};
#pragma unroll 1
  for (int k = 0; k < HID; k += 64) {
    v16bf a0 = load_a_frag(h_bf, HID, m0, k);
    v16bf b0 = load_b_frag(wmhT, HID, n0, k);
    acc2a = wmma_bf16(a0, b0, acc2a);
    v16bf a1 = load_a_frag(h_bf, HID, m0, k + 32);
    v16bf b1 = load_b_frag(wmhT, HID, n0, k + 32);
    acc2b = wmma_bf16(a1, b1, acc2b);
  }
  // x @ wmx (K = 64)
  v8f acc1 = {};
#pragma unroll
  for (int k = 0; k < NEMBD; k += 32) {
    v16bf a = load_a_frag(x, NEMBD, m0, k);
    v16bf b = load_b_frag(wmxT, NEMBD, n0, k);
    acc1 = wmma_bf16(a, b, acc1);
  }

  const int col   = n0 + (l & 15);
  const int rbase = m0 + ((l >> 4) << 3);
#pragma unroll
  for (int r = 0; r < 8; ++r)
    m_bf[(size_t)(rbase + r) * HID + col] =
        (bf16)(acc1[r] * (acc2a[r] + acc2b[r]));
}

// ---- per-step kernel 2: z = x@wx + m@wh + b, gates, state update -----------
// Each wave owns one 16x16 tile in "gate-column" space j and accumulates all
// four gates (4 independent WMMA chains sharing the A fragment).
__global__ __launch_bounds__(256)
void step_z_kernel(const bf16* __restrict__ words, const bf16* __restrict__ m_bf,
                   const bf16* __restrict__ wxT, const bf16* __restrict__ whT,
                   const float* __restrict__ bias, const float* __restrict__ Mmask,
                   float* __restrict__ c_ws, float* __restrict__ h_ws,
                   bf16* __restrict__ h_bf, bf16* __restrict__ hs_bf,
                   float* __restrict__ cells_out, float* __restrict__ states_out,
                   int t) {
  const int wave = threadIdx.x >> 5;
  const int l    = threadIdx.x & 31;
  const int m0   = (wave & 1) * 16;
  const int j0   = blockIdx.x * 64 + (wave >> 1) * 16;
  const bf16* x  = words + (size_t)t * B_SZ * NEMBD;

  v8f acc[4] = {v8f{}, v8f{}, v8f{}, v8f{}};
#pragma unroll 1
  for (int k = 0; k < HID; k += 32) {
    v16bf a = load_a_frag(m_bf, HID, m0, k);
#pragma unroll
    for (int gi = 0; gi < 4; ++gi) {
      v16bf b = load_b_frag(whT, HID, gi * HID + j0, k);
      acc[gi] = wmma_bf16(a, b, acc[gi]);
    }
  }
#pragma unroll
  for (int k = 0; k < NEMBD; k += 32) {
    v16bf a = load_a_frag(x, NEMBD, m0, k);
#pragma unroll
    for (int gi = 0; gi < 4; ++gi) {
      v16bf b = load_b_frag(wxT, NEMBD, gi * HID + j0, k);
      acc[gi] = wmma_bf16(a, b, acc[gi]);
    }
  }

  const int jj    = j0 + (l & 15);
  const int rbase = m0 + ((l >> 4) << 3);
  const float bi  = bias[jj];
  const float bff = bias[HID + jj];
  const float bo  = bias[2 * HID + jj];
  const float bu  = bias[3 * HID + jj];
#pragma unroll
  for (int r = 0; r < 8; ++r) {
    const int bidx = rbase + r; // batch row
    const float ig = 1.f / (1.f + __expf(-(acc[0][r] + bi)));
    const float fg = 1.f / (1.f + __expf(-(acc[1][r] + bff)));
    const float og = 1.f / (1.f + __expf(-(acc[2][r] + bo)));
    const float ug = tanhf(acc[3][r] + bu);
    const size_t sidx = (size_t)bidx * HID + jj;
    const float c_old = c_ws[sidx];
    const float h_old = h_ws[sidx];
    const float ct = fg * c_old + ig * ug;
    const float ht = og * tanhf(ct);
    const float mt = Mmask[bidx * T_SZ + t];
    const float c_new = ct * mt + c_old * (1.f - mt);
    const float h_new = ht * mt + h_old * (1.f - mt);
    c_ws[sidx] = c_new;
    h_ws[sidx] = h_new;
    h_bf[sidx] = (bf16)h_new;
    cells_out[(size_t)t * B_SZ * HID + sidx] = c_new;          // cells[t,b,j]
    hs_bf[((size_t)bidx * T_SZ + t) * HID + jj] = (bf16)h_new; // batch-major hs
    if (t == T_SZ - 1) {
      states_out[sidx] = c_new;                       // states[0] = c_final
      states_out[(size_t)B_SZ * HID + sidx] = h_new;  // states[1] = h_final
    }
  }
}

// ---- logits: [B*T, NVOCAB] = hs_bf @ w_out + b_out -------------------------
__global__ __launch_bounds__(256)
void logits_kernel(const bf16* __restrict__ hs_bf, const bf16* __restrict__ woutT,
                   const float* __restrict__ b_out, float* __restrict__ logits) {
  const int wave = threadIdx.x >> 5;
  const int l    = threadIdx.x & 31;
  const int m0   = blockIdx.x * 32 + (wave & 1) * 16;
  const int n0   = blockIdx.y * 64 + (wave >> 1) * 16;

  v8f acca = {}, accb = {};
#pragma unroll 1
  for (int k = 0; k < HID; k += 64) {
    v16bf a0 = load_a_frag(hs_bf, HID, m0, k);
    v16bf b0 = load_b_frag(woutT, HID, n0, k);
    acca = wmma_bf16(a0, b0, acca);
    v16bf a1 = load_a_frag(hs_bf, HID, m0, k + 32);
    v16bf b1 = load_b_frag(woutT, HID, n0, k + 32);
    accb = wmma_bf16(a1, b1, accb);
  }
  const int col   = n0 + (l & 15);
  const int rbase = m0 + ((l >> 4) << 3);
  const float bo  = b_out[col];
#pragma unroll
  for (int r = 0; r < 8; ++r)
    logits[(size_t)(rbase + r) * NVOCAB + col] = acca[r] + accb[r] + bo;
}

// ---------------------------------------------------------------------------
extern "C" void kernel_launch(void* const* d_in, const int* in_sizes, int n_in,
                              void* d_out, int out_size, void* d_ws, size_t ws_size,
                              hipStream_t stream) {
  (void)in_sizes; (void)n_in; (void)out_size; (void)ws_size;
  const int*   X     = (const int*)d_in[0];
  const float* Mmask = (const float*)d_in[1];
  const float* S     = (const float*)d_in[2];
  const float* embd  = (const float*)d_in[3];
  const float* wx    = (const float*)d_in[4];
  const float* wh    = (const float*)d_in[5];
  const float* wmx   = (const float*)d_in[6];
  const float* wmh   = (const float*)d_in[7];
  const float* bias  = (const float*)d_in[8];
  const float* gx    = (const float*)d_in[9];
  const float* gh    = (const float*)d_in[10];
  const float* gmx   = (const float*)d_in[11];
  const float* gmh   = (const float*)d_in[12];
  const float* w_out = (const float*)d_in[13];
  const float* b_out = (const float*)d_in[14];

  // Workspace carve-out (~191 MB total)
  char* ws = (char*)d_ws;
  size_t off = 0;
  auto carve = [&](size_t bytes) -> void* {
    void* p = ws + off;
    off = (off + bytes + 255) & ~(size_t)255;
    return p;
  };
  bf16*  whT   = (bf16*)carve((size_t)ZDIM * HID * sizeof(bf16));      // 128 MB
  bf16*  wmhT  = (bf16*)carve((size_t)HID * HID * sizeof(bf16));       //  32 MB
  bf16*  wxT   = (bf16*)carve((size_t)ZDIM * NEMBD * sizeof(bf16));    //   2 MB
  bf16*  wmxT  = (bf16*)carve((size_t)HID * NEMBD * sizeof(bf16));     // 0.5 MB
  bf16*  woutT = (bf16*)carve((size_t)NVOCAB * HID * sizeof(bf16));    //   2 MB
  bf16*  words = (bf16*)carve((size_t)T_SZ * B_SZ * NEMBD * sizeof(bf16));
  bf16*  m_bf  = (bf16*)carve((size_t)B_SZ * HID * sizeof(bf16));
  bf16*  h_bf  = (bf16*)carve((size_t)B_SZ * HID * sizeof(bf16));
  bf16*  hs_bf = (bf16*)carve((size_t)B_SZ * T_SZ * HID * sizeof(bf16)); // 16 MB
  float* c_ws  = (float*)carve((size_t)B_SZ * HID * sizeof(float));
  float* h_ws  = (float*)carve((size_t)B_SZ * HID * sizeof(float));

  float* cells_out  = (float*)d_out;                                   // [T,B,HID]
  float* states_out = cells_out + (size_t)T_SZ * B_SZ * HID;           // [2,B,HID]
  float* logits_out = states_out + (size_t)2 * B_SZ * HID;             // [B*T,NVOCAB]

  // One-time prep (single HBM pass over fp32 weights)
  wn_transpose_kernel<<<(ZDIM + 255) / 256, 256, 0, stream>>>(wh, gh, whT, HID, ZDIM, 1);
  wn_transpose_kernel<<<(HID + 255) / 256, 256, 0, stream>>>(wmh, gmh, wmhT, HID, HID, 1);
  wn_transpose_kernel<<<(ZDIM + 255) / 256, 256, 0, stream>>>(wx, gx, wxT, NEMBD, ZDIM, 1);
  wn_transpose_kernel<<<(HID + 255) / 256, 256, 0, stream>>>(wmx, gmx, wmxT, NEMBD, HID, 1);
  wn_transpose_kernel<<<(NVOCAB + 255) / 256, 256, 0, stream>>>(w_out, nullptr, woutT, HID, NVOCAB, 0);
  embed_kernel<<<(T_SZ * B_SZ * NEMBD + 255) / 256, 256, 0, stream>>>(X, embd, words);
  init_state_kernel<<<(B_SZ * HID + 255) / 256, 256, 0, stream>>>(S, c_ws, h_ws, h_bf);

  // Sequential recurrence: weights stream from L2 (160 MB bf16 fits in 192 MB)
  for (int t = 0; t < T_SZ; ++t) {
    step_m_kernel<<<HID / 64, 256, 0, stream>>>(words, h_bf, wmxT, wmhT, m_bf, t);
    step_z_kernel<<<HID / 64, 256, 0, stream>>>(words, m_bf, wxT, whT, bias, Mmask,
                                                c_ws, h_ws, h_bf, hs_bf,
                                                cells_out, states_out, t);
  }

  // Final projection
  logits_kernel<<<dim3((B_SZ * T_SZ) / 32, NVOCAB / 64), 256, 0, stream>>>(
      hs_bf, woutT, b_out, logits_out);
}
